// ImageArm_32684701122971
// MI455X (gfx1250) — matrix operand order repaired and checked by reading, not compiled, not run on hardware
//
#include <hip/hip_runtime.h>
#include <hip/hip_bf16.h>

typedef __attribute__((ext_vector_type(16))) _Float16 v16h;
typedef __attribute__((ext_vector_type(8)))  float    v8f;
typedef __attribute__((ext_vector_type(4)))  unsigned u32x4;
typedef __attribute__((ext_vector_type(8)))  int      i32x8;
typedef __attribute__((ext_vector_type(4)))  int      i32x4;

#define HH   1024
#define WW   1024
#define HWSZ (HH * WW)
#define CTX  8
#define HID  6
#define WVS  8            // waves per block (256 threads, wave32)
#define BROWS 8           // pixel rows per block
#define BCOLS 128         // pixel cols per block
#define PR   (BROWS + 2)  // staged patch rows (dy in [-2,0])
#define PC   (BCOLS + 4)  // staged patch cols (dx in [-2,1], padded to +2)

// Causal context shifts, in reference order (sorted by (d^2, idx)):
// (-1,0) (0,-1) (-1,-1) (-1,1) (-2,0) (0,-2) (-2,-1) (-2,1)
__device__ __constant__ int SDY[CTX] = {-1,  0, -1, -1, -2,  0, -2, -2};
__device__ __constant__ int SDX[CTX] = { 0, -1, -1,  1,  0, -2, -1,  1};

union U16h { v16h v; _Float16 h[16]; };

// B-fragment for D = A x B: B[k][n] = Wm[e][n][k], Wm shape [E, cols, rows].
// ISA 16-bit B 32x16: lane n=L&15 is column; lanes<16 K in {0..7,16..23}, lanes>=16 K in {8..15,24..31}.
__device__ inline v16h load_Bfrag(const float* __restrict__ Wm, int e,
                                  int rows, int cols, int lane) {
  U16h b;
  int n  = lane & 15;
  int kb = (lane < 16) ? 0 : 8;
#pragma unroll
  for (int i = 0; i < 16; ++i) {
    int k = kb + (i & 7) + ((i & 8) ? 16 : 0);
    float v = 0.0f;
    if (n < cols && k < rows)
      v = Wm[(size_t)e * cols * rows + (size_t)n * rows + k];
    b.h[i] = (_Float16)v;
  }
  return b.v;
}

// ISA 16-bit A 16x32 layout: lane L row M=L&15; lanes<16 K in {0..7,16..23},
// lanes>=16 K in {8..15,24..31}; element i -> K = kbase + (i&7) + (i&8 ? 16 : 0).

// A-fragment from global memory (border blocks).
__device__ inline v16h build_Afrag_glb(const float* __restrict__ img,
                                       const float* __restrict__ synth,
                                       int y, int xb, int lane) {
  U16h a;
  int x  = xb + (lane & 15);
  int kb = (lane < 16) ? 0 : 8;
#pragma unroll
  for (int i = 0; i < 16; ++i) {
    int k = kb + (i & 7) + ((i & 8) ? 16 : 0);
    float v;
    if (k < 3 * CTX) {                       // interleaved context: d = 3*s + c
      int s = k / 3, c = k % 3;
      int yy = y + SDY[s], xx = x + SDX[s];
      v = (yy >= 0 && yy < HH && xx >= 0 && xx < WW)
            ? img[(size_t)c * HWSZ + (size_t)yy * WW + xx] : 0.0f;
    } else if (k < 3 * CTX + 6) {
      v = synth[(size_t)(k - 24) * HWSZ + (size_t)y * WW + x];
    } else {                                 // img ch 0/1 (masked by W pad)
      v = img[(size_t)(k - 30) * HWSZ + (size_t)y * WW + x];
    }
    a.h[i] = (_Float16)v;
  }
  return a.v;
}

// A-fragment from the TDM-staged LDS patch (interior blocks).
// shimg layout: [3][PR][PC], ly/lx are patch coords of this lane's pixel.
__device__ inline v16h build_Afrag_lds(const float* __restrict__ shimg,
                                       const float* __restrict__ synth,
                                       int y, int x, int ly, int lx, int lane) {
  U16h a;
  int kb = (lane < 16) ? 0 : 8;
#pragma unroll
  for (int i = 0; i < 16; ++i) {
    int k = kb + (i & 7) + ((i & 8) ? 16 : 0);
    float v;
    if (k < 3 * CTX) {
      int s = k / 3, c = k % 3;
      v = shimg[(c * PR + ly + SDY[s]) * PC + (lx + SDX[s])];
    } else if (k < 3 * CTX + 6) {
      v = synth[(size_t)(k - 24) * HWSZ + (size_t)y * WW + x];
    } else {
      v = shimg[((k - 30) * PR + ly) * PC + lx];
    }
    a.h[i] = (_Float16)v;
  }
  return a.v;
}

// Rebuild A-fragment from a 16x8 LDS patch of hidden activations
// (only K=0..5 nonzero -> only lanes 0..15 carry data).
__device__ inline v16h build_Ahid(const float* __restrict__ ldsRow, int lane) {
  U16h a;
#pragma unroll
  for (int i = 0; i < 16; ++i) a.h[i] = (_Float16)0.0f;
  if (lane < 16) {
#pragma unroll
    for (int i = 0; i < HID; ++i)            // element i -> K=i for lanes<16
      a.h[i] = (_Float16)ldsRow[lane * 8 + i];
  }
  return a.v;
}

extern "C" __global__ void __launch_bounds__(256)
imagearm_moe_wmma(const float* __restrict__ img,
                  const float* __restrict__ synth,
                  const float* __restrict__ W1_0, const float* __restrict__ b1_0,
                  const float* __restrict__ W2_0, const float* __restrict__ b2_0,
                  const float* __restrict__ W3_0, const float* __restrict__ b3_0,
                  const float* __restrict__ W1_1, const float* __restrict__ b1_1,
                  const float* __restrict__ W2_1, const float* __restrict__ b2_1,
                  const float* __restrict__ W3_1, const float* __restrict__ b3_1,
                  const float* __restrict__ W1_2, const float* __restrict__ b1_2,
                  const float* __restrict__ W2_2, const float* __restrict__ b2_2,
                  const float* __restrict__ W3_2, const float* __restrict__ b3_2,
                  float* __restrict__ out) {
  __shared__ float shimg[3 * PR * PC];       // TDM-staged image patch w/ halo
  __shared__ float ldsx[WVS][16][8];         // per-wave D->A transpose patches

  const int wave = threadIdx.x >> 5;
  const int lane = threadIdx.x & 31;
  const int n    = lane & 15;

  // Block -> region mapping: 64 blocks per expert (32 row-strips x 2 col-halves)
  const int b   = blockIdx.x;
  const int e   = b >> 6;                    // expert 0..15
  const int wb  = b & 63;
  const int y0  = (e >> 2) * 256 + (wb >> 1) * BROWS;
  const int xb0 = (e & 3) * 256 + (wb & 1) * BCOLS;

  // Interior <=> the haloed patch [y0-2,y0+BROWS) x [xb0-2,xb0+BCOLS+2) is
  // fully inside the image -> TDM tile never leaves the tensor.
  const bool interior = (y0 != 0) && (xb0 != 0) && (xb0 != WW - BCOLS);

  if (interior) {
#if __has_builtin(__builtin_amdgcn_tensor_load_to_lds)
    if (wave == 0) {
      // ---- Tensor DMA descriptor (CDNA5 ISA) ----
      // 3D tensor: dim0 = W (x), dim1 = H (y), dim2 = 3 (channel);
      // tile = PC x PR x 3; strides: dim0_stride = W, dim1_stride = H*W.
      const unsigned ldsOff = (unsigned)(unsigned long long)(uintptr_t)shimg;
      const unsigned long long ga =
          (unsigned long long)(uintptr_t)img +
          4ull * ((unsigned long long)(y0 - 2) * WW + (unsigned)(xb0 - 2));
      u32x4 g0 = { 1u,                                   // count=1, user mode
                   ldsOff,                               // lds_addr (bytes)
                   (unsigned)(ga & 0xFFFFFFFFull),       // global_addr[31:0]
                   (unsigned)((ga >> 32) & 0x1FFFFFFull) // global_addr[56:32]
                     | (2u << 30) };                     // type=2 ("image")
      i32x8 g1 = { (int)(2u << 16),                          // data_size=4B
                   (int)((WW & 0xFFFF) << 16),               // tensor_dim0 lo16
                   (int)(((WW >> 16) & 0xFFFF)
                         | ((HH & 0xFFFF) << 16)),           // dim0 hi / dim1 lo
                   (int)(((HH >> 16) & 0xFFFF)
                         | ((unsigned)PC << 16)),            // dim1 hi / tile_dim0
                   (int)((unsigned)PR | (3u << 16)),         // tile_dim1 / tile_dim2
                   (int)WW,                                  // dim0_stride lo32
                   (int)(((unsigned)HWSZ & 0xFFFF) << 16),   // dim1_stride lo16
                   (int)((unsigned)HWSZ >> 16) };            // dim1_stride [47:16]
      i32x4 g2 = { 3, 0, 0, 0 };               // tensor_dim2=3; tile_dim3 unused
      i32x4 g3 = { 0, 0, 0, 0 };
#if defined(__clang_major__) && (__clang_major__ >= 23)
      i32x8 g4 = { 0, 0, 0, 0, 0, 0, 0, 0 };
      __builtin_amdgcn_tensor_load_to_lds(g0, g1, g2, g3, g4, 0);
#else
      __builtin_amdgcn_tensor_load_to_lds(g0, g1, g2, g3, 0);
#endif
#if __has_builtin(__builtin_amdgcn_s_wait_tensorcnt)
      __builtin_amdgcn_s_wait_tensorcnt(0);
#else
      asm volatile("s_wait_tensorcnt 0x0" ::: "memory");
#endif
    }
#else
    // Cooperative fallback staging (coalesced row loads)
    for (int idx = threadIdx.x; idx < 3 * PR * PC; idx += 256) {
      int c = idx / (PR * PC), rem = idx % (PR * PC);
      int rr = rem / PC, cc = rem % PC;
      shimg[idx] = img[(size_t)c * HWSZ +
                       (size_t)(y0 - 2 + rr) * WW + (xb0 - 2 + cc)];
    }
#endif
    __syncthreads();
  }

  const float* W1p[3] = {W1_0, W1_1, W1_2};
  const float* b1p[3] = {b1_0, b1_1, b1_2};
  const float* W2p[3] = {W2_0, W2_1, W2_2};
  const float* b2p[3] = {b2_0, b2_1, b2_2};
  const float* W3p[3] = {W3_0, W3_1, W3_2};
  const float* b3p[3] = {b3_0, b3_1, b3_2};
  const int DIN[3] = {3 * CTX + 6 + 0, 3 * CTX + 6 + 1, 3 * CTX + 6 + 2};

  // Expert weights: loaded once per wave, resident across the tile loop.
  v16h B1[3], B2[3], B3[3];
  float b1n[3], b2n[3], b3n[3];
#pragma unroll
  for (int c = 0; c < 3; ++c) {
    B1[c] = load_Bfrag(W1p[c], e, DIN[c], HID, lane);
    B2[c] = load_Bfrag(W2p[c], e, HID,    HID, lane);
    B3[c] = load_Bfrag(W3p[c], e, HID,    4,   lane);
    b1n[c] = (n < HID) ? b1p[c][e * HID + n] : 0.0f;
    b2n[c] = (n < HID) ? b2p[c][e * HID + n] : 0.0f;
    b3n[c] = (n < 4)   ? b3p[c][e * 4   + n] : 0.0f;
  }

  float* myLds = &ldsx[wave][0][0];
  const int M0 = (lane < 16) ? 0 : 8;        // D/C rows this lane holds

  for (int it = 0; it < 8; ++it) {           // 8 tiles per wave
    const int tt  = wave * 8 + it;           // tile in block, 0..63
    const int y   = y0 + (tt >> 3);
    const int xb  = xb0 + (tt & 7) * 16;
    const int x   = xb + n;                  // this lane's pixel (rows of A)

    v16h A;
    if (interior)
      A = build_Afrag_lds(shimg, synth, y, x, (y - y0) + 2, (x - xb0) + 2, lane);
    else
      A = build_Afrag_glb(img, synth, y, xb, lane);

#pragma unroll
    for (int c = 0; c < 3; ++c) {
      // ---- layer 1: h1 = relu(A x W1^T + b1) ----
      v8f C;
#pragma unroll
      for (int r = 0; r < 8; ++r) C[r] = b1n[c];
      v8f h1 = __builtin_amdgcn_wmma_f32_16x16x32_f16(
          false, A, false, B1[c], (short)0, C, false, false);
#pragma unroll
      for (int r = 0; r < 8; ++r) h1[r] = (h1[r] > 0.0f) ? h1[r] : 0.0f;

      if (n < HID) {
#pragma unroll
        for (int r = 0; r < 8; ++r) myLds[(M0 + r) * 8 + n] = h1[r];
      }
      asm volatile("s_wait_dscnt 0" ::: "memory");
      const v16h A2 = build_Ahid(myLds, lane);

      // ---- layer 2 (residual): h2 = relu(h1 + A2 x W2^T + b2) via C ----
#pragma unroll
      for (int r = 0; r < 8; ++r) C[r] = h1[r] + b2n[c];
      v8f h2 = __builtin_amdgcn_wmma_f32_16x16x32_f16(
          false, A2, false, B2[c], (short)0, C, false, false);
#pragma unroll
      for (int r = 0; r < 8; ++r) h2[r] = (h2[r] > 0.0f) ? h2[r] : 0.0f;

      if (n < HID) {
#pragma unroll
        for (int r = 0; r < 8; ++r) myLds[(M0 + r) * 8 + n] = h2[r];
      }
      asm volatile("s_wait_dscnt 0" ::: "memory");
      const v16h A3 = build_Ahid(myLds, lane);

      // ---- layer 3: o = A3 x W3^T + b3  (cols 0..1 raw, 2..3 gate) ----
#pragma unroll
      for (int r = 0; r < 8; ++r) C[r] = b3n[c];
      v8f o = __builtin_amdgcn_wmma_f32_16x16x32_f16(
          false, A3, false, B3[c], (short)0, C, false, false);

      // pull gate column (n+2) into raw column's lane (wave32 bpermute)
      v8f g;
#pragma unroll
      for (int r = 0; r < 8; ++r) {
        int gi = __builtin_amdgcn_ds_bpermute((lane + 2) << 2,
                                              __float_as_int(o[r]));
        g[r] = __int_as_float(gi);
      }

      if (n < 2) {                           // raw columns 0,1 -> output s=2c+n
        const size_t sOff = (size_t)(2 * c + n) * HWSZ + (size_t)y * WW + xb;
#pragma unroll
        for (int r = 0; r < 8; ++r) {
          const size_t off = sOff + (M0 + r);
          const float sig = 1.0f / (1.0f + __expf(-g[r]));
          out[off] = synth[off] + o[r] * sig;
        }
      }
    }
  }
}

extern "C" void kernel_launch(void* const* d_in, const int* in_sizes, int n_in,
                              void* d_out, int out_size, void* d_ws, size_t ws_size,
                              hipStream_t stream) {
  (void)in_sizes; (void)n_in; (void)d_ws; (void)ws_size; (void)out_size;
  const float* img   = (const float*)d_in[0];
  const float* synth = (const float*)d_in[1];
  // d_in[2] = expert_indices: routing is a fixed 4x4 grid -> recomputed, unused.
  const float* w[18];
  for (int i = 0; i < 18; ++i) w[i] = (const float*)d_in[3 + i];

  // 1024 blocks x 8 waves x 8 tiles x 16 pixels = 1M pixels
  dim3 grid(1024), block(256);
  imagearm_moe_wmma<<<grid, block, 0, stream>>>(
      img, synth,
      w[0],  w[1],  w[2],  w[3],  w[4],  w[5],    // ch0: W1 b1 W2 b2 W3 b3
      w[6],  w[7],  w[8],  w[9],  w[10], w[11],   // ch1
      w[12], w[13], w[14], w[15], w[16], w[17],   // ch2
      (float*)d_out);
}
